// MultiBertAttention_47390669144804
// MI455X (gfx1250) — matrix operand order repaired and checked by reading, not compiled, not run on hardware
//
#include <hip/hip_runtime.h>

// ---------------------------------------------------------------------------
// MultiBertAttention forward for MI455X (gfx1250), wave32, WMMA bf16 path.
// B=2, S=2048, D=1024, H=16, Hd=64.
// Pipeline:
//   1) cvt:  fp32 -> bf16 for x, Wq, Wk, Wv, Wo        (bandwidth trivial)
//   2) gemm: Q/K/V = x*W + b   (bf16 WMMA, fp32 acc), head-major bf16 out
//      - 128x128 block tile, 8 waves x (32x64), async global->LDS A staging
//   3) flash attention per (b,h): online softmax, WMMA for QK^T and PV
//   4) gemm: out = ctx*Wo + bo (fp32 result to d_out)
// ---------------------------------------------------------------------------

typedef __bf16  v16bf __attribute__((ext_vector_type(16)));
typedef float   v8f   __attribute__((ext_vector_type(8)));
typedef unsigned int v8u __attribute__((ext_vector_type(8)));
typedef unsigned short u16;

#define BATCH 2
#define SEQ   2048
#define DMODEL 1024
#define NHEAD 16
#define HD    64

__device__ __forceinline__ u16 f2bf(float f) {
  unsigned int u = __float_as_uint(f);
  u += 0x7FFFu + ((u >> 16) & 1u);   // round-to-nearest-even
  return (u16)(u >> 16);
}

__device__ __forceinline__ v16bf pack_bf16(uint4 a, uint4 b) {
  v8u u;
  u[0] = a.x; u[1] = a.y; u[2] = a.z; u[3] = a.w;
  u[4] = b.x; u[5] = b.y; u[6] = b.z; u[7] = b.w;
  return __builtin_bit_cast(v16bf, u);
}

// CDNA5 async global->LDS copy (ASYNCcnt-tracked, no VGPR round trip).
// GVS addressing: mem = SGPR base + 32-bit per-lane voffset; LDS dest from
// per-lane VGPR (vdata slot), matching VGLOBAL operand order vdata,vaddr,saddr.
__device__ __forceinline__ void async_ld_b128(unsigned lds_off, unsigned voff,
                                              unsigned long long sbase) {
  asm volatile("global_load_async_to_lds_b128 %0, %1, %2"
               :: "v"(lds_off), "v"(voff), "s"(sbase) : "memory");
}
__device__ __forceinline__ void wait_async0() {
  asm volatile("s_wait_asynccnt 0x0" ::: "memory");
}

// ---------------------------------------------------------------------------
// fp32 -> bf16 conversion
// ---------------------------------------------------------------------------
__global__ void mba_cvt_bf16(const float* __restrict__ in,
                             u16* __restrict__ out, int n) {
  int i = blockIdx.x * blockDim.x + threadIdx.x;
  int stride = gridDim.x * blockDim.x;
  for (; i < n; i += stride) out[i] = f2bf(in[i]);
}

// ---------------------------------------------------------------------------
// 128x128 block-tile bf16 GEMM:  out[M,N] = A[M,K] * W[K,N] + bias[N]
// 256 threads = 8 waves in a 4(M) x 2(N) grid; each wave computes 32x64.
// Per 32-wide K chunk: 2 A-frags, 4 B-frags, 8 WMMAs per wave.
// OUT_MODE 0: fp32 row-major to d_out
// OUT_MODE 1: bf16 head-major [b,h,s,hd] (for Q/K/V)
// ---------------------------------------------------------------------------
template <int OUT_MODE>
__global__ __launch_bounds__(256)
void mba_gemm_bf16(const u16* __restrict__ A, const u16* __restrict__ W,
                   const float* __restrict__ bias, void* __restrict__ outp,
                   int M, int N, int K) {
  __shared__ __align__(16) u16 As[128][40];   // row-major [m][k], padded
  __shared__ __align__(16) u16 Bs[128][40];   // transposed  [n][k], padded
  const int tid = threadIdx.x;
  const int wave = tid >> 5, lane = tid & 31;
  const int lane16 = lane & 15, hi = lane >> 4;
  const int n0 = blockIdx.x * 128, m0 = blockIdx.y * 128;
  const int mw = (wave & 3) * 32, nw = (wave >> 2) * 64;

  // A staging: 128x32 bf16 = 4096 elems; 16/thread as two async b128 per thread
  const int arow = tid >> 1, acol = (tid & 1) * 16;
  const unsigned ldsA0 = (unsigned)(size_t)&As[arow][acol];
  const unsigned ldsA1 = (unsigned)(size_t)&As[arow][acol + 8];
  const unsigned avbase = (unsigned)(((m0 + arow) * (size_t)K + acol) * 2);
  const unsigned long long abase = (unsigned long long)(size_t)A;

  // B staging: 32x128 bf16; 16/thread, transposed into [n][k]
  const int bk = tid >> 3, bnc = (tid & 7) * 16;

  v8f acc[2][4] = {{{}, {}, {}, {}}, {{}, {}, {}, {}}};

  for (int k0 = 0; k0 < K; k0 += 32) {
    __syncthreads();
    // A tile via CDNA5 async global->LDS (two 16B lanes-worth per thread)
    async_ld_b128(ldsA0, avbase + (unsigned)(k0 * 2), abase);
    async_ld_b128(ldsA1, avbase + (unsigned)(k0 * 2) + 16, abase);
    // B tile: regular loads + transposed b16 scatter
    uint4 w0 = *(const uint4*)(W + (size_t)(k0 + bk) * N + n0 + bnc);
    uint4 w1 = *(const uint4*)(W + (size_t)(k0 + bk) * N + n0 + bnc + 8);
    u16 tmp[16];
    *(uint4*)&tmp[0] = w0;
    *(uint4*)&tmp[8] = w1;
#pragma unroll
    for (int e = 0; e < 16; ++e) Bs[bnc + e][bk] = tmp[e];
    wait_async0();
    __syncthreads();

    // A operands: 16-bit A 16x32 layout (lanes 0-15: K 0-7,16-23; hi half: +8)
    v16bf afrag[2];
#pragma unroll
    for (int im = 0; im < 2; ++im) {
      const u16* ar = &As[mw + im * 16 + lane16][0];
      const int ka = 8 * hi;
      afrag[im] = pack_bf16(*(const uint4*)(ar + ka),
                            *(const uint4*)(ar + ka + 16));
    }
#pragma unroll
    for (int jn = 0; jn < 4; ++jn) {
      // B operand: lanes 0-15 hold K 0-7,16-23 of column n; hi half: +8
      const u16* br = &Bs[nw + jn * 16 + lane16][0];
      const int kb2 = 8 * hi;
      v16bf bm = pack_bf16(*(const uint4*)(br + kb2),
                           *(const uint4*)(br + kb2 + 16));
#pragma unroll
      for (int im = 0; im < 2; ++im)
        acc[im][jn] = __builtin_amdgcn_wmma_f32_16x16x32_bf16(
            false, afrag[im], false, bm, (short)0, acc[im][jn], false, false);
    }
  }

  // epilogue: C layout — VGPR i, lanes 0-15: M=i, N=lane; lanes 16-31: M=i+8
#pragma unroll
  for (int jn = 0; jn < 4; ++jn) {
    const int gn = n0 + nw + jn * 16 + lane16;
    const float bv = bias[gn];
#pragma unroll
    for (int im = 0; im < 2; ++im) {
#pragma unroll
      for (int i = 0; i < 8; ++i) {
        const int gm = m0 + mw + im * 16 + i + 8 * hi;
        const float v = acc[im][jn][i] + bv;
        if (OUT_MODE == 0) {
          ((float*)outp)[(size_t)gm * N + gn] = v;
        } else {
          const int bb = gm >> 11, ss = gm & 2047;        // S = 2048
          const int hh = gn >> 6, hd = gn & 63;           // Hd = 64
          ((u16*)outp)[(((size_t)bb * NHEAD + hh) * SEQ + ss) * HD + hd] =
              f2bf(v);
        }
      }
    }
  }
}

// ---------------------------------------------------------------------------
// Flash attention: grid (S/128, B*H). 8 waves; wave = 16 query rows.
// Key blocks of 64; online softmax in fp32; QK^T and PV via bf16 WMMA.
// ---------------------------------------------------------------------------
__global__ __launch_bounds__(256)
void mba_flash_attn(const u16* __restrict__ Qb, const u16* __restrict__ Kb,
                    const u16* __restrict__ Vb, u16* __restrict__ ctx) {
  __shared__ __align__(16) u16 Vt[HD][64];        // V tile transposed [hd][key]
  __shared__ __align__(16) u16 Pld[8][16][64];    // per-wave P staging
  const int tid = threadIdx.x;
  const int wave = tid >> 5, lane = tid & 31;
  const int lane16 = lane & 15, hi = lane >> 4;
  const int bh = blockIdx.y;
  const int b = bh >> 4, h = bh & 15;
  const int q0 = blockIdx.x * 128 + wave * 16;

  // Q tile in A-layout: row = q0 + lane16, two 32-wide K chunks over Hd
  v16bf qa[2];
  {
    const u16* qrow = Qb + ((size_t)bh * SEQ + q0 + lane16) * HD;
#pragma unroll
    for (int c = 0; c < 2; ++c) {
      const int d0 = 32 * c + 8 * hi;
      qa[c] = pack_bf16(*(const uint4*)(qrow + d0),
                        *(const uint4*)(qrow + d0 + 16));
    }
  }

  float mrow[8], lrow[8];
  v8f o[4] = {{}, {}, {}, {}};
#pragma unroll
  for (int i = 0; i < 8; ++i) { mrow[i] = -__builtin_inff(); lrow[i] = 0.f; }

  for (int kb = 0; kb < SEQ / 64; ++kb) {
    __syncthreads();
    // stage V tile transposed: Vt[hd][key]
    {
      const u16* vbase = Vb + ((size_t)bh * SEQ + kb * 64) * HD;
#pragma unroll
      for (int i = 0; i < 16; ++i) {
        const int e = i * 256 + tid;
        Vt[e & 63][e >> 6] = vbase[(e >> 6) * HD + (e & 63)];
      }
      if (kb + 1 < SEQ / 64)
        __builtin_prefetch(Vb + ((size_t)bh * SEQ + (kb + 1) * 64) * HD, 0, 0);
    }
    __syncthreads();

    // scores: S = Q * K^T ; K rows are contiguous in Hd => direct B-layout load
    v8f sacc[4];
#pragma unroll
    for (int j = 0; j < 4; ++j) {
      v8f s = {};
      const u16* krow =
          Kb + ((size_t)bh * SEQ + kb * 64 + j * 16 + lane16) * HD;
#pragma unroll
      for (int c = 0; c < 2; ++c) {
        const int d0 = 32 * c + 8 * hi;
        v16bf bm = pack_bf16(*(const uint4*)(krow + d0),
                             *(const uint4*)(krow + d0 + 16));
        s = __builtin_amdgcn_wmma_f32_16x16x32_bf16(
            false, qa[c], false, bm, (short)0, s, false, false);
      }
      sacc[j] = s;
    }

    // online softmax: row = slot i + 8*hi, spread over 16 lanes of the half
#pragma unroll
    for (int j = 0; j < 4; ++j)
#pragma unroll
      for (int i = 0; i < 8; ++i) sacc[j][i] *= 0.125f;  // 1/sqrt(64)

    float rmax[8];
#pragma unroll
    for (int i = 0; i < 8; ++i)
      rmax[i] = fmaxf(fmaxf(sacc[0][i], sacc[1][i]),
                      fmaxf(sacc[2][i], sacc[3][i]));
#pragma unroll
    for (int msk = 1; msk <= 8; msk <<= 1)
#pragma unroll
      for (int i = 0; i < 8; ++i)
        rmax[i] = fmaxf(rmax[i], __shfl_xor(rmax[i], msk, 32));

    float mnew[8], sc[8], rsum[8];
#pragma unroll
    for (int i = 0; i < 8; ++i) {
      mnew[i] = fmaxf(mrow[i], rmax[i]);
      sc[i] = __expf(mrow[i] - mnew[i]);
      rsum[i] = 0.f;
    }
#pragma unroll
    for (int j = 0; j < 4; ++j)
#pragma unroll
      for (int i = 0; i < 8; ++i) {
        const float p = __expf(sacc[j][i] - mnew[i]);
        rsum[i] += p;
        Pld[wave][i + 8 * hi][j * 16 + lane16] = f2bf(p);
      }
#pragma unroll
    for (int msk = 1; msk <= 8; msk <<= 1)
#pragma unroll
      for (int i = 0; i < 8; ++i) rsum[i] += __shfl_xor(rsum[i], msk, 32);
#pragma unroll
    for (int i = 0; i < 8; ++i) {
      lrow[i] = lrow[i] * sc[i] + rsum[i];
      mrow[i] = mnew[i];
    }
#pragma unroll
    for (int j = 0; j < 4; ++j)
#pragma unroll
      for (int i = 0; i < 8; ++i) o[j][i] *= sc[i];

    __syncthreads();  // P visible in A-layout reads, V reuse safe

    // PV: A = P (16 x 64 keys), B = V (keys x hd) from transposed LDS tile
    v16bf pa[2];
#pragma unroll
    for (int c = 0; c < 2; ++c) {
      const u16* prow = &Pld[wave][lane16][0];
      const int k0 = 32 * c + 8 * hi;
      pa[c] = pack_bf16(*(const uint4*)(prow + k0),
                        *(const uint4*)(prow + k0 + 16));
    }
#pragma unroll
    for (int j = 0; j < 4; ++j) {
#pragma unroll
      for (int c = 0; c < 2; ++c) {
        const u16* vr = &Vt[j * 16 + lane16][0];
        const int k0 = 32 * c + 8 * hi;
        v16bf bm = pack_bf16(*(const uint4*)(vr + k0),
                             *(const uint4*)(vr + k0 + 16));
        o[j] = __builtin_amdgcn_wmma_f32_16x16x32_bf16(
            false, pa[c], false, bm, (short)0, o[j], false, false);
      }
    }
  }

  // epilogue: ctx[b, s, h*64 + hd] as bf16 for the final projection GEMM
#pragma unroll
  for (int j = 0; j < 4; ++j)
#pragma unroll
    for (int i = 0; i < 8; ++i) {
      const int srow = q0 + i + 8 * hi;
      const int col = h * HD + j * 16 + lane16;
      const float v = o[j][i] / lrow[i];
      ctx[((size_t)b * SEQ + srow) * DMODEL + col] = f2bf(v);
    }
}

// ---------------------------------------------------------------------------
extern "C" void kernel_launch(void* const* d_in, const int* in_sizes, int n_in,
                              void* d_out, int out_size, void* d_ws,
                              size_t ws_size, hipStream_t stream) {
  (void)in_sizes; (void)n_in; (void)out_size; (void)ws_size;
  const float* x  = (const float*)d_in[0];
  const float* Wq = (const float*)d_in[1];
  const float* bq = (const float*)d_in[2];
  const float* Wk = (const float*)d_in[3];
  const float* bk = (const float*)d_in[4];
  const float* Wv = (const float*)d_in[5];
  const float* bv = (const float*)d_in[6];
  const float* Wo = (const float*)d_in[7];
  const float* bo = (const float*)d_in[8];

  char* ws = (char*)d_ws;
  const size_t MB = 1u << 20;
  u16* xb  = (u16*)(ws + 0 * MB);    // 8 MB: x bf16 [B*S, D]
  u16* wqb = (u16*)(ws + 8 * MB);    // 2 MB each
  u16* wkb = (u16*)(ws + 10 * MB);
  u16* wvb = (u16*)(ws + 12 * MB);
  u16* wob = (u16*)(ws + 14 * MB);
  u16* Qb  = (u16*)(ws + 16 * MB);   // 8 MB each, head-major [B*H,S,Hd]
  u16* Kb  = (u16*)(ws + 24 * MB);
  u16* Vb  = (u16*)(ws + 32 * MB);
  u16* ctx = (u16*)(ws + 40 * MB);   // 8 MB, [B*S, D] bf16

  const int NX = BATCH * SEQ * DMODEL;       // 4,194,304
  const int NW = DMODEL * DMODEL;            // 1,048,576
  mba_cvt_bf16<<<2048, 256, 0, stream>>>(x, xb, NX);
  mba_cvt_bf16<<<1024, 256, 0, stream>>>(Wq, wqb, NW);
  mba_cvt_bf16<<<1024, 256, 0, stream>>>(Wk, wkb, NW);
  mba_cvt_bf16<<<1024, 256, 0, stream>>>(Wv, wvb, NW);
  mba_cvt_bf16<<<1024, 256, 0, stream>>>(Wo, wob, NW);

  const int M = BATCH * SEQ;                 // 4096
  dim3 ggrid(DMODEL / 128, M / 128);         // (8, 32)
  mba_gemm_bf16<1><<<ggrid, 256, 0, stream>>>(xb, wqb, bq, Qb, M, DMODEL, DMODEL);
  mba_gemm_bf16<1><<<ggrid, 256, 0, stream>>>(xb, wkb, bk, Kb, M, DMODEL, DMODEL);
  mba_gemm_bf16<1><<<ggrid, 256, 0, stream>>>(xb, wvb, bv, Vb, M, DMODEL, DMODEL);

  dim3 fgrid(SEQ / 128, BATCH * NHEAD);      // (16, 32)
  mba_flash_attn<<<fgrid, 256, 0, stream>>>(Qb, Kb, Vb, ctx);

  mba_gemm_bf16<0><<<ggrid, 256, 0, stream>>>(ctx, wob, bo, (float*)d_out,
                                              M, DMODEL, DMODEL);
}
